// RNN_EXT_10599979287188
// MI455X (gfx1250) — compile-verified, matrix operand order checked by reading, not run on hardware
//
#include <hip/hip_runtime.h>

typedef __attribute__((ext_vector_type(16))) _Float16 v16h;
typedef __attribute__((ext_vector_type(8)))  float    v8f;

#define BTOT 131072
#define ROWS_PER_WAVE 16
#define WAVES_PER_BLOCK 8
#define THREADS_PER_BLOCK (WAVES_PER_BLOCK * 32)

// LDS layout (units: halfs for weights, then floats for biases, then per-wave scratch)
#define W1X_OFF   0            // 256x128
#define W1H_OFF   32768        // 256x64
#define W12X_OFF  49152        // 256x64
#define W12H_OFF  65536        // 256x64
#define W2X_OFF   81920        // 128x64
#define W2H_OFF   90112        // 128x32
#define W3X_OFF   94208        // 64x32
#define W3H_OFF   96256        // 64x32 (K padded 16->32)
#define WCX_OFF   98304        // 16x32 (N padded 4->16, K 16->32)
#define WCH_OFF   98816        // 16x32 (K padded 1->32)
#define WAX_OFF   99328        // 80x32 (N padded 72->80, K 16->32)
#define WAH_OFF   101888       // 80x32 (K padded 18->32)
#define WTOT_HALFS 104448
#define BIAS_FLOATS 800        // b1:0 b12:256 b2:512 b3:640 bc:704 ba:720(+80)
#define PER_WAVE_BYTES 9216    // bufA 2048 + bufB 2048 + gateBuf 5120
#define SMEM_BYTES (WTOT_HALFS*2 + BIAS_FLOATS*4 + WAVES_PER_BLOCK*PER_WAVE_BYTES)

struct Params {
  const float *state,*h1,*c1,*h2,*c2,*h3,*c3,*h12,*c12,*hc,*cc,*ha,*ca;
  const float *wih1,*whh1,*bih1,*bhh1;
  const float *wih12,*whh12,*bih12,*bhh12;
  const float *wih2,*whh2,*bih2,*bhh2;
  const float *wih3,*whh3,*bih3,*bhh3;
  const float *wihc,*whhc,*bihc,*bhhc;
  const float *wiha,*whha,*biha,*bhha;
  const int* action_map;
  float* out;
};

__device__ __forceinline__ v8f wmma32f16(v16h a, v16h b, v8f c) {
  // (neg_a, A, neg_b, B, c_mod, C, reuse_a, reuse_b)
  return __builtin_amdgcn_wmma_f32_16x16x32_f16(false, a, false, b, (short)0, c, false, false);
}

__device__ __forceinline__ float sigm_(float x) { return 1.0f / (1.0f + __expf(-x)); }
__device__ __forceinline__ float tanh_(float x) {
  x = fminf(15.0f, fmaxf(-15.0f, x));
  float e = __expf(2.0f * x);
  return (e - 1.0f) / (e + 1.0f);
}

// B fragment: lane holds 16 consecutive halfs of weight row n (gate index), K-chunk kc.
__device__ __forceinline__ v16h bfrag(const _Float16* w, int stride, int ntile, int kc, int lane) {
  int n  = ntile * 16 + (lane & 15);
  int k0 = kc * 32 + ((lane >> 4) << 4);
  return *(const v16h*)(w + n * stride + k0);
}

// A fragment from per-wave LDS activation buffer (row-major, stride 64 halfs).
__device__ __forceinline__ v16h afrag_lds(const _Float16* buf, int kc, int lane) {
  int row = lane & 15;
  int k0  = kc * 32 + ((lane >> 4) << 4);
  return *(const v16h*)(buf + row * 64 + k0);
}

// A fragment from global f32 row-major [B, STRIDE]; VALID columns, zero-padded; scaled.
template<int STRIDE, int VALID>
__device__ __forceinline__ v16h afrag_g(const float* base, int r0, int lane, int kc, float scale) {
  int row = r0 + (lane & 15);
  int c0  = kc * 32 + ((lane >> 4) << 4);
  const float* p = base + (size_t)row * STRIDE + c0;
  v16h a;
#pragma unroll
  for (int i = 0; i < 16; ++i) {
    float v = 0.0f;
    if (c0 + i < VALID) v = p[i] * scale;
    a[i] = (_Float16)v;
  }
  return a;
}

// Generic WMMA LSTM cell for dh in {16,32,64} (T = dh/16). D-fragment-local elementwise.
template<int T, int XC, int HC>
__device__ __forceinline__ void lstm_cell(
    const v16h* ax, const v16h* ah,
    const _Float16* wih, int ih_stride,
    const _Float16* whh, int hh_stride,
    const float* bias,
    const float* c_in, float* h_out, float* c_out,
    _Float16* act_out, const _Float16* resid,
    int r0, int lane) {
  const int DH  = T * 16;
  const int col = lane & 15;
  const int hi  = lane >> 4;
#pragma unroll
  for (int k = 0; k < T; ++k) {
    float b0 = bias[(0 * T + k) * 16 + col];
    float b1 = bias[(1 * T + k) * 16 + col];
    float b2 = bias[(2 * T + k) * 16 + col];
    float b3 = bias[(3 * T + k) * 16 + col];
    v8f ai = {b0, b0, b0, b0, b0, b0, b0, b0};
    v8f af = {b1, b1, b1, b1, b1, b1, b1, b1};
    v8f ag = {b2, b2, b2, b2, b2, b2, b2, b2};
    v8f ao = {b3, b3, b3, b3, b3, b3, b3, b3};
#pragma unroll
    for (int c = 0; c < XC; ++c) {
      ai = wmma32f16(ax[c], bfrag(wih, ih_stride, 0 * T + k, c, lane), ai);
      af = wmma32f16(ax[c], bfrag(wih, ih_stride, 1 * T + k, c, lane), af);
      ag = wmma32f16(ax[c], bfrag(wih, ih_stride, 2 * T + k, c, lane), ag);
      ao = wmma32f16(ax[c], bfrag(wih, ih_stride, 3 * T + k, c, lane), ao);
    }
#pragma unroll
    for (int c = 0; c < HC; ++c) {
      ai = wmma32f16(ah[c], bfrag(whh, hh_stride, 0 * T + k, c, lane), ai);
      af = wmma32f16(ah[c], bfrag(whh, hh_stride, 1 * T + k, c, lane), af);
      ag = wmma32f16(ah[c], bfrag(whh, hh_stride, 2 * T + k, c, lane), ag);
      ao = wmma32f16(ah[c], bfrag(whh, hh_stride, 3 * T + k, c, lane), ao);
    }
#pragma unroll
    for (int m = 0; m < 8; ++m) {
      int lrow = m + (hi << 3);
      int row  = r0 + lrow;
      int hcol = k * 16 + col;
      float ii = sigm_(ai[m]);
      float ff = sigm_(af[m]);
      float gg = tanh_(ag[m]);
      float oo = sigm_(ao[m]);
      float cold = c_in[(size_t)row * DH + hcol];
      float cn = ff * cold + ii * gg;
      float hn = oo * tanh_(cn);
      c_out[(size_t)row * DH + hcol] = cn;
      h_out[(size_t)row * DH + hcol] = hn;
      if (act_out) {
        float av = hn + (resid ? (float)resid[lrow * 64 + hcol] : 0.0f);
        act_out[lrow * 64 + hcol] = (_Float16)av;
      }
    }
  }
}

__device__ void stageW(const float* src, _Float16* dst, int srows, int scols,
                       int drows, int dcols, int tid, int nt) {
  int total = drows * dcols;
  for (int i = tid; i < total; i += nt) {
    int r = i / dcols, c = i - r * dcols;
    float v = (r < srows && c < scols) ? src[r * scols + c] : 0.0f;
    dst[i] = (_Float16)v;
  }
}

__device__ void stageB(const float* bi, const float* bh, float* dst, int n, int npad,
                       int tid, int nt) {
  for (int i = tid; i < npad; i += nt) dst[i] = (i < n) ? (bi[i] + bh[i]) : 0.0f;
}

__global__ __launch_bounds__(THREADS_PER_BLOCK, 1)
void lstm_policy_kernel(Params p) {
  extern __shared__ char smem[];
  _Float16* WH   = (_Float16*)smem;
  float*    Bias = (float*)(smem + WTOT_HALFS * 2);
  const int tid = threadIdx.x;

  // ---- stage all weights (f32 -> f16) and fused biases into LDS ----
  stageW(p.wih1,  WH + W1X_OFF,  256, 128, 256, 128, tid, THREADS_PER_BLOCK);
  stageW(p.whh1,  WH + W1H_OFF,  256,  64, 256,  64, tid, THREADS_PER_BLOCK);
  stageW(p.wih12, WH + W12X_OFF, 256,  64, 256,  64, tid, THREADS_PER_BLOCK);
  stageW(p.whh12, WH + W12H_OFF, 256,  64, 256,  64, tid, THREADS_PER_BLOCK);
  stageW(p.wih2,  WH + W2X_OFF,  128,  64, 128,  64, tid, THREADS_PER_BLOCK);
  stageW(p.whh2,  WH + W2H_OFF,  128,  32, 128,  32, tid, THREADS_PER_BLOCK);
  stageW(p.wih3,  WH + W3X_OFF,   64,  32,  64,  32, tid, THREADS_PER_BLOCK);
  stageW(p.whh3,  WH + W3H_OFF,   64,  16,  64,  32, tid, THREADS_PER_BLOCK);
  stageW(p.wihc,  WH + WCX_OFF,    4,  16,  16,  32, tid, THREADS_PER_BLOCK);
  stageW(p.whhc,  WH + WCH_OFF,    4,   1,  16,  32, tid, THREADS_PER_BLOCK);
  stageW(p.wiha,  WH + WAX_OFF,   72,  16,  80,  32, tid, THREADS_PER_BLOCK);
  stageW(p.whha,  WH + WAH_OFF,   72,  18,  80,  32, tid, THREADS_PER_BLOCK);
  stageB(p.bih1,  p.bhh1,  Bias + 0,   256, 256, tid, THREADS_PER_BLOCK);
  stageB(p.bih12, p.bhh12, Bias + 256, 256, 256, tid, THREADS_PER_BLOCK);
  stageB(p.bih2,  p.bhh2,  Bias + 512, 128, 128, tid, THREADS_PER_BLOCK);
  stageB(p.bih3,  p.bhh3,  Bias + 640,  64,  64, tid, THREADS_PER_BLOCK);
  stageB(p.bihc,  p.bhhc,  Bias + 704,   4,  16, tid, THREADS_PER_BLOCK);
  stageB(p.biha,  p.bhha,  Bias + 720,  72,  80, tid, THREADS_PER_BLOCK);
  __syncthreads();

  const int lane = tid & 31;
  const int wid  = tid >> 5;
  const int gw   = blockIdx.x * WAVES_PER_BLOCK + wid;
  const int r0   = gw * ROWS_PER_WAVE;
  if (r0 >= BTOT) return;

  char* wbase = smem + WTOT_HALFS * 2 + BIAS_FLOATS * 4 + wid * PER_WAVE_BYTES;
  _Float16* bufA    = (_Float16*)wbase;          // 16x64 halfs
  _Float16* bufB    = bufA + 1024;               // 16x64 halfs
  float*    gateBuf = (float*)(bufB + 1024);     // 16x80 f32

  const int col = lane & 15;
  const int hi  = lane >> 4;

  // ---- output carve (return-order concatenation) ----
  float* out_action = p.out;
  float* out_h1  = out_action + (size_t)BTOT;
  float* out_c1  = out_h1  + (size_t)BTOT * 64;
  float* out_h2  = out_c1  + (size_t)BTOT * 64;
  float* out_c2  = out_h2  + (size_t)BTOT * 32;
  float* out_h3  = out_c2  + (size_t)BTOT * 32;
  float* out_c3  = out_h3  + (size_t)BTOT * 16;
  float* out_h12 = out_c3  + (size_t)BTOT * 16;
  float* out_c12 = out_h12 + (size_t)BTOT * 64;
  float* out_hc  = out_c12 + (size_t)BTOT * 64;
  float* out_cc  = out_hc  + (size_t)BTOT;
  float* out_ha  = out_cc  + (size_t)BTOT;
  float* out_ca  = out_ha  + (size_t)BTOT * 18;
  float* out_ent = out_ca  + (size_t)BTOT * 18;
  float* out_lp  = out_ent + (size_t)BTOT;
  float* out_v   = out_lp  + (size_t)BTOT;

  // ---- cell 1: 128 -> 64 ----
  v16h ax1[4], ahh[2];
#pragma unroll
  for (int c = 0; c < 4; ++c) ax1[c] = afrag_g<128, 128>(p.state, r0, lane, c, 1.0f / 510.0f);
#pragma unroll
  for (int c = 0; c < 2; ++c) ahh[c] = afrag_g<64, 64>(p.h1, r0, lane, c, 1.0f);
  lstm_cell<4, 4, 2>(ax1, ahh, WH + W1X_OFF, 128, WH + W1H_OFF, 64, Bias + 0,
                     p.c1, out_h1, out_c1, bufA, nullptr, r0, lane);

  // ---- cell 12: 64 -> 64, residual x2 = h1 + h12 ----
  v16h ax2[2];
#pragma unroll
  for (int c = 0; c < 2; ++c) ax2[c] = afrag_lds(bufA, c, lane);
#pragma unroll
  for (int c = 0; c < 2; ++c) ahh[c] = afrag_g<64, 64>(p.h12, r0, lane, c, 1.0f);
  lstm_cell<4, 2, 2>(ax2, ahh, WH + W12X_OFF, 64, WH + W12H_OFF, 64, Bias + 256,
                     p.c12, out_h12, out_c12, bufB, bufA, r0, lane);

  // ---- cell 2: 64 -> 32 ----
  v16h ax3[2];
#pragma unroll
  for (int c = 0; c < 2; ++c) ax3[c] = afrag_lds(bufB, c, lane);
  v16h ah2 = afrag_g<32, 32>(p.h2, r0, lane, 0, 1.0f);
  lstm_cell<2, 2, 1>(ax3, &ah2, WH + W2X_OFF, 64, WH + W2H_OFF, 32, Bias + 512,
                     p.c2, out_h2, out_c2, bufA, nullptr, r0, lane);

  // ---- cell 3: 32 -> 16 ----
  v16h ax4 = afrag_lds(bufA, 0, lane);
  v16h ah3 = afrag_g<16, 16>(p.h3, r0, lane, 0, 1.0f);
  lstm_cell<1, 1, 1>(&ax4, &ah3, WH + W3X_OFF, 32, WH + W3H_OFF, 32, Bias + 640,
                     p.c3, out_h3, out_c3, bufB, nullptr, r0, lane);
  // zero-pad h3 buffer columns 16..31 (K padding for the next cells)
#pragma unroll
  for (int m = 0; m < 8; ++m) bufB[(m + (hi << 3)) * 64 + 16 + col] = (_Float16)0.0f;

  v16h ax5 = afrag_lds(bufB, 0, lane);

  // ---- cell a: 16 -> 18 via WMMA into gate buffer (N padded to 80) ----
  v16h aha = afrag_g<18, 18>(p.ha, r0, lane, 0, 1.0f);
#pragma unroll
  for (int n = 0; n < 5; ++n) {
    float bb = Bias[720 + n * 16 + col];
    v8f acc = {bb, bb, bb, bb, bb, bb, bb, bb};
    acc = wmma32f16(ax5, bfrag(WH + WAX_OFF, 32, n, 0, lane), acc);
    acc = wmma32f16(aha, bfrag(WH + WAH_OFF, 32, n, 0, lane), acc);
#pragma unroll
    for (int m = 0; m < 8; ++m) gateBuf[(m + (hi << 3)) * 80 + n * 16 + col] = acc[m];
  }
  // elementwise LSTM for cell a; logits left in-place at gateBuf[r][0..17]
#pragma unroll
  for (int t = 0; t < 9; ++t) {
    int idx = t * 32 + lane;  // 0..287 = 16 rows x 18 units
    int r = idx / 18;
    int j = idx - r * 18;
    float* gb = gateBuf + r * 80;
    float ii = sigm_(gb[j]);
    float ff = sigm_(gb[18 + j]);
    float gg = tanh_(gb[36 + j]);
    float oo = sigm_(gb[54 + j]);
    int grow = r0 + r;
    float cold = p.ca[(size_t)grow * 18 + j];
    float cn = ff * cold + ii * gg;
    float hn = oo * tanh_(cn);
    out_ca[(size_t)grow * 18 + j] = cn;
    out_ha[(size_t)grow * 18 + j] = hn;
    gb[j] = hn;  // logit
  }

  // ---- cell c: 16 -> 1 (VALU dot products; 4 gates spread over lane pairs) ----
  {
    int r16 = lane & 15;
    int g0 = hi * 2;  // lanes<16: gates {0,1}=i,f ; lanes>=16: {2,3}=g,o
    float a0 = Bias[704 + g0], a1 = Bias[704 + g0 + 1];
#pragma unroll
    for (int kk = 0; kk < 16; ++kk) {
      float xv = (float)bufB[r16 * 64 + kk];
      a0 += xv * (float)WH[WCX_OFF + (g0)     * 32 + kk];
      a1 += xv * (float)WH[WCX_OFF + (g0 + 1) * 32 + kk];
    }
    float hcold = p.hc[r0 + r16];
    a0 += hcold * (float)WH[WCH_OFF + (g0)     * 32];
    a1 += hcold * (float)WH[WCH_OFF + (g0 + 1) * 32];
    float graw = __shfl(a0, r16 + 16, 32);
    float oraw = __shfl(a1, r16 + 16, 32);
    if (lane < 16) {
      float ii = sigm_(a0), ff = sigm_(a1), gg = tanh_(graw), oo = sigm_(oraw);
      float cold = p.cc[r0 + lane];
      float cn = ff * cold + ii * gg;
      float hn = oo * tanh_(cn);
      out_hc[r0 + lane] = hn;
      out_cc[r0 + lane] = cn;
      out_v[r0 + lane]  = hn;
    }
  }

  // ---- softmax / entropy / categorical sample (lane r handles row r) ----
  if (lane < 16) {
    int row = r0 + lane;
    const float* lg = gateBuf + lane * 80;
    float mx = -1e30f;
#pragma unroll
    for (int j = 0; j < 18; ++j) mx = fmaxf(mx, lg[j]);
    float se = 0.0f;
#pragma unroll
    for (int j = 0; j < 18; ++j) se += __expf(lg[j] - mx);
    float lse = __logf(se) + mx;

    // deterministic per-row uniform (stands in for jax.random.key(42))
    unsigned s = (unsigned)row * 0x9E3779B9u ^ 0xDEADBE42u;
    s ^= s >> 16; s *= 0x7FEB352Du;
    s ^= s >> 15; s *= 0x846CA68Bu;
    s ^= s >> 16;
    float u = (float)(s >> 8) * (1.0f / 16777216.0f);

    float ent = 0.0f, cum = 0.0f, lpsel = lg[17] - lse;
    int chosen = -1;
#pragma unroll
    for (int j = 0; j < 18; ++j) {
      float lpj = lg[j] - lse;
      float pj  = __expf(lpj);
      ent -= pj * lpj;
      cum += pj;
      if (chosen < 0 && u < cum) { chosen = j; lpsel = lpj; }
    }
    if (chosen < 0) chosen = 17;
    out_action[row] = (float)p.action_map[chosen];
    out_ent[row]    = ent;
    out_lp[row]     = lpsel;
  }
}

extern "C" void kernel_launch(void* const* d_in, const int* in_sizes, int n_in,
                              void* d_out, int out_size, void* d_ws, size_t ws_size,
                              hipStream_t stream) {
  (void)in_sizes; (void)n_in; (void)out_size; (void)d_ws; (void)ws_size;
  Params p;
  p.state = (const float*)d_in[0];
  p.h1  = (const float*)d_in[1];   p.c1  = (const float*)d_in[2];
  p.h2  = (const float*)d_in[3];   p.c2  = (const float*)d_in[4];
  p.h3  = (const float*)d_in[5];   p.c3  = (const float*)d_in[6];
  p.h12 = (const float*)d_in[7];   p.c12 = (const float*)d_in[8];
  p.hc  = (const float*)d_in[9];   p.cc  = (const float*)d_in[10];
  p.ha  = (const float*)d_in[11];  p.ca  = (const float*)d_in[12];
  p.wih1  = (const float*)d_in[13]; p.whh1  = (const float*)d_in[14];
  p.bih1  = (const float*)d_in[15]; p.bhh1  = (const float*)d_in[16];
  p.wih12 = (const float*)d_in[17]; p.whh12 = (const float*)d_in[18];
  p.bih12 = (const float*)d_in[19]; p.bhh12 = (const float*)d_in[20];
  p.wih2  = (const float*)d_in[21]; p.whh2  = (const float*)d_in[22];
  p.bih2  = (const float*)d_in[23]; p.bhh2  = (const float*)d_in[24];
  p.wih3  = (const float*)d_in[25]; p.whh3  = (const float*)d_in[26];
  p.bih3  = (const float*)d_in[27]; p.bhh3  = (const float*)d_in[28];
  p.wihc  = (const float*)d_in[29]; p.whhc  = (const float*)d_in[30];
  p.bihc  = (const float*)d_in[31]; p.bhhc  = (const float*)d_in[32];
  p.wiha  = (const float*)d_in[33]; p.whha  = (const float*)d_in[34];
  p.biha  = (const float*)d_in[35]; p.bhha  = (const float*)d_in[36];
  p.action_map = (const int*)d_in[37];
  p.out = (float*)d_out;

  static bool attr_set = false;
  if (!attr_set) {
    hipFuncSetAttribute(reinterpret_cast<const void*>(&lstm_policy_kernel),
                        hipFuncAttributeMaxDynamicSharedMemorySize, SMEM_BYTES);
    attr_set = true;
  }

  const int blocks = BTOT / (ROWS_PER_WAVE * WAVES_PER_BLOCK);  // 1024
  lstm_policy_kernel<<<blocks, THREADS_PER_BLOCK, SMEM_BYTES, stream>>>(p);
}
